// VectorQuantizer_12807592477166
// MI455X (gfx1250) — matrix operand order repaired and checked by reading, not compiled, not run on hardware
//
#include <hip/hip_runtime.h>

// Problem constants (from reference): z [32,256,64,64] f32, codebook [512,256] f32
#define ZB    32
#define ZC    256
#define ZHW   4096            // 64*64
#define NPIX  131072          // 32*4096
#define NCODE 512
#define KSTEP 32              // channels per bf16 WMMA K-step (== staged chunk)
#define BPITCH_U32 20         // u32 per code row in LDS (32 bf16 = 16 u32, +4 pad -> 20-bank stride)

typedef __attribute__((ext_vector_type(16))) __bf16 v16bf;
typedef __attribute__((ext_vector_type(8)))  float  v8f;

// ---------------------------------------------------------------------------
// Kernel 0: cnorm[k] = ||codebook[k]||^2 (exact fp32) ; zero the loss accumulator
// ---------------------------------------------------------------------------
__global__ void __launch_bounds__(256) vq_init_kernel(const float* __restrict__ cbk,
                                                      float* __restrict__ cnorm,
                                                      float* __restrict__ lossAcc) {
  const int k = blockIdx.x * 256 + threadIdx.x;   // 0..511
  if (k < NCODE) {
    const float4* r = (const float4*)(cbk + (size_t)k * ZC);
    float s = 0.f;
#pragma unroll 8
    for (int i = 0; i < ZC / 4; ++i) {
      float4 v = r[i];
      s += v.x * v.x + v.y * v.y + v.z * v.z + v.w * v.w;
    }
    cnorm[k] = s;
  }
  if (k == 0) lossAcc[0] = 0.f;
}

// ---------------------------------------------------------------------------
// Kernel 1: split-precision bf16x3 WMMA score GEMM + fused argmin.
// Block = 512 threads = 16 waves; 64 rows/block.
// Wave w: mtile = w>>2 (16 rows), code-set = w&3 (128 codes = 8 subtiles).
// score(z,c) ~= zh*ch + zl*ch + zh*cl  via V_WMMA_F32_16X16X32_BF16.
// ---------------------------------------------------------------------------
__global__ void __launch_bounds__(512) vq_argmin_kernel(const float* __restrict__ z,
                                                        const float* __restrict__ cbk,
                                                        const float* __restrict__ cnorm,
                                                        float* __restrict__ oidx) {
  // Codebook chunk (512 codes x 32 channels), hi and lo bf16, K-contiguous rows.
  __shared__ __align__(16) unsigned bhS[NCODE * BPITCH_U32];   // 40,960 B
  __shared__ __align__(16) unsigned blS[NCODE * BPITCH_U32];   // 40,960 B
  __shared__ float cnormS[NCODE];
  __shared__ float redD[4 * 16 * 4];
  __shared__ int   redC[4 * 16 * 4];

  const int tid    = threadIdx.x;
  const int lane   = tid & 31;
  const int lanelo = lane & 15;
  const int lanehi = lane >> 4;
  const int w      = tid >> 5;
  const int mt     = w >> 2;
  const int cs     = w & 3;
  const int codeBase = cs * 128;

  const int row   = blockIdx.x * 64 + mt * 16 + lanelo;   // tiles never straddle batch
  const int b     = row >> 12;
  const int hw    = row & 4095;
  const int zbase = (b << 20) + hw;                        // (b*256)<<12 + hw

  if (tid < NCODE) cnormS[tid] = cnorm[tid];

  v8f acc[8] = {};   // 16 rows x 128 codes, fp32 scores

  for (int c0 = 0; c0 < ZC; c0 += KSTEP) {
    __syncthreads();
    // ---- Stage codebook[*, c0..c0+31] as bf16 hi/lo into LDS -------------
    // pair p: code=p>>4, kp=p&15 -> channels c0+2kp, c0+2kp+1 (coalesced float2
    // reads; packed b32 LDS writes are bank-conflict-free: stride 1, +20/code).
#pragma unroll
    for (int i = 0; i < 16; ++i) {
      int p    = tid + i * 512;        // 0..8191
      int code = p >> 4;
      int kp   = p & 15;
      float2 f = *(const float2*)(cbk + code * ZC + c0 + kp * 2);
      __bf16 h0 = (__bf16)f.x;
      __bf16 h1 = (__bf16)f.y;
      __bf16 l0 = (__bf16)(f.x - (float)h0);
      __bf16 l1 = (__bf16)(f.y - (float)h1);
      union { unsigned u; __bf16 h[2]; } ph, pl;
      ph.h[0] = h0; ph.h[1] = h1;
      pl.h[0] = l0; pl.h[1] = l1;
      bhS[code * BPITCH_U32 + kp] = ph.u;
      blS[code * BPITCH_U32 + kp] = pl.u;
    }
    __syncthreads();

    // ---- A fragment: z rows, 16 channels per lane (contiguous K map:
    // element j <-> K = j + 16*lanehi), then split hi/lo. -------
    float fz[16];
#pragma unroll
    for (int j = 0; j < 16; ++j) {
      int K = j + 16 * lanehi;
      fz[j] = z[zbase + ((c0 + K) << 12)];
    }
    v16bf ah, al;
#pragma unroll
    for (int j = 0; j < 16; ++j) {
      __bf16 h = (__bf16)fz[j];
      ah[j] = h;
      al[j] = (__bf16)(fz[j] - (float)h);
    }

    // ---- 8 code subtiles x 3 bf16 GEMM terms ----------------------------
    // B 32x16 map: lane holds code=codeBase+s*16+lanelo, K=16*lanehi..+15
    // contiguous -> two ds_load_b128 per fragment.
#pragma unroll
    for (int s = 0; s < 8; ++s) {
      int code = codeBase + s * 16 + lanelo;
      const uint4* bhp = (const uint4*)(bhS + code * BPITCH_U32 + lanehi * 8);
      const uint4* blp = (const uint4*)(blS + code * BPITCH_U32 + lanehi * 8);
      union { uint4 q[2]; v16bf v; } ubh, ubl;
      ubh.q[0] = bhp[0]; ubh.q[1] = bhp[1];
      ubl.q[0] = blp[0]; ubl.q[1] = blp[1];
      // correction terms first, dominant term last
      acc[s] = __builtin_amdgcn_wmma_f32_16x16x32_bf16(
          false, al, false, ubh.v, (short)0, acc[s], false, false);
      acc[s] = __builtin_amdgcn_wmma_f32_16x16x32_bf16(
          false, ah, false, ubl.v, (short)0, acc[s], false, false);
      acc[s] = __builtin_amdgcn_wmma_f32_16x16x32_bf16(
          false, ah, false, ubh.v, (short)0, acc[s], false, false);
    }
  }

  // dist = cnorm - 2*score; per-lane min over 8 subtiles, then across the
  // 16-lane half. Tie-break: lowest code (matches jnp.argmin first-occurrence).
#pragma unroll
  for (int v = 0; v < 8; ++v) {
    float bd = 3.4e38f;
    int   bc = 0x7fffffff;
#pragma unroll
    for (int s = 0; s < 8; ++s) {
      int   code = codeBase + s * 16 + lanelo;
      float d    = cnormS[code] - 2.0f * acc[s][v];
      if (d < bd || (d == bd && code < bc)) { bd = d; bc = code; }
    }
#pragma unroll
    for (int off = 1; off < 16; off <<= 1) {
      float od = __shfl_xor(bd, off, 32);
      int   oc = __shfl_xor(bc, off, 32);
      if (od < bd || (od == bd && oc < bc)) { bd = od; bc = oc; }
    }
    if (lanelo == 0) {
      int m = v + 8 * lanehi;
      redD[(mt * 16 + m) * 4 + cs] = bd;
      redC[(mt * 16 + m) * 4 + cs] = bc;
    }
  }
  __syncthreads();

  if (tid < 64) {   // one thread per row: reduce across the 4 code-set waves
    float bd = 3.4e38f;
    int   bc = 0x7fffffff;
#pragma unroll
    for (int j = 0; j < 4; ++j) {
      float d = redD[tid * 4 + j];
      int   c = redC[tid * 4 + j];
      if (d < bd || (d == bd && c < bc)) { bd = d; bc = c; }
    }
    oidx[blockIdx.x * 64 + tid] = (float)bc;
  }
}

// ---------------------------------------------------------------------------
// Kernel 2: z_q scatter to [B,C,H,W] + squared-error accumulation
// ---------------------------------------------------------------------------
__global__ void __launch_bounds__(256) vq_gather_kernel(const float* __restrict__ z,
                                                        const float* __restrict__ cbk,
                                                        const float* __restrict__ oidx,
                                                        float* __restrict__ zq,
                                                        float* __restrict__ lossAcc) {
  __shared__ float red[256];
  const int tid  = threadIdx.x;
  const int n    = blockIdx.x * 256 + tid;
  const int code = (int)(oidx[n] + 0.5f);
  const int b    = n >> 12;
  const int hw   = n & 4095;
  const float4* crow = (const float4*)(cbk + (size_t)code * ZC);

  float local = 0.f;
#pragma unroll 4
  for (int c4 = 0; c4 < ZC / 4; ++c4) {
    float4 q    = crow[c4];
    int    base = (b << 20) + ((c4 * 4) << 12) + hw;   // coalesced across lanes
    float z0 = z[base];
    float z1 = z[base + (1 << 12)];
    float z2 = z[base + (2 << 12)];
    float z3 = z[base + (3 << 12)];
    zq[base]             = q.x;
    zq[base + (1 << 12)] = q.y;
    zq[base + (2 << 12)] = q.z;
    zq[base + (3 << 12)] = q.w;
    float d0 = q.x - z0, d1 = q.y - z1, d2 = q.z - z2, d3 = q.w - z3;
    local += d0 * d0 + d1 * d1 + d2 * d2 + d3 * d3;
  }

  red[tid] = local;
  __syncthreads();
  for (int s = 128; s > 0; s >>= 1) {
    if (tid < s) red[tid] += red[tid + s];
    __syncthreads();
  }
  if (tid == 0) atomicAdd(lossAcc, red[0]);
}

// ---------------------------------------------------------------------------
// Kernel 3: loss = (1 + BETA) * SSE / numel   (both loss terms equal numerically)
// ---------------------------------------------------------------------------
__global__ void vq_final_kernel(const float* __restrict__ lossAcc, float* __restrict__ oloss) {
  oloss[0] = 1.25f * lossAcc[0] / 33554432.0f;
}

// ---------------------------------------------------------------------------
extern "C" void kernel_launch(void* const* d_in, const int* in_sizes, int n_in,
                              void* d_out, int out_size, void* d_ws, size_t ws_size,
                              hipStream_t stream) {
  const float* z   = (const float*)d_in[0];   // [32,256,64,64]
  const float* cbk = (const float*)d_in[1];   // [512,256]

  float* out   = (float*)d_out;
  float* zq    = out;                                  // 33,554,432 floats [B,C,H,W]
  float* oidx  = out + (size_t)NPIX * ZC;              // 131,072 entries
  float* oloss = oidx + NPIX;                          // 1 scalar

  float* cnorm   = (float*)d_ws;                       // 512 floats
  float* lossAcc = cnorm + NCODE;                      // 1 float

  vq_init_kernel<<<2, 256, 0, stream>>>(cbk, cnorm, lossAcc);
  vq_argmin_kernel<<<NPIX / 64, 512, 0, stream>>>(z, cbk, cnorm, oidx);
  vq_gather_kernel<<<NPIX / 256, 256, 0, stream>>>(z, cbk, oidx, zq, lossAcc);
  vq_final_kernel<<<1, 1, 0, stream>>>(lossAcc, oloss);
}
